// triplane_fea_21199958573495
// MI455X (gfx1250) — matrix operand order repaired and checked by reading, not compiled, not run on hardware
//
#include <hip/hip_runtime.h>

typedef __attribute__((ext_vector_type(16))) _Float16 v16h;
typedef __attribute__((ext_vector_type(8)))  _Float16 v8h;
typedef __attribute__((ext_vector_type(8)))  float    v8f;

#define WAVES 8
#define R_GRID 256

// ---- workspace layout (halves) ----
// wt1 [112][128] @0, wt2 [112][128] @14336, wt3 [64][96] @28672,
// cw1 [64][64] @34816, cw2 [64][64] @38912, cw3 [16][64] @43008  -> 44032 halves
// then float biases: b1[112] b2[112] b3[64] cb1[64] cb2[64] cb3[16] = 432 floats
#define H_TOTAL 44032
#define F_TOTAL 432

__device__ __forceinline__ int iminx(int a, int b) { return a < b ? a : b; }
__device__ __forceinline__ int imaxx(int a, int b) { return a > b ? a : b; }

// ---------------- weight/bias prep: f32 -> padded f16 ----------------
__global__ void triplane_prep_kernel(const float* __restrict__ w1, const float* __restrict__ b1,
                                     const float* __restrict__ w2, const float* __restrict__ b2,
                                     const float* __restrict__ w3, const float* __restrict__ b3,
                                     const float* __restrict__ cw1, const float* __restrict__ cb1,
                                     const float* __restrict__ cw2, const float* __restrict__ cb2,
                                     const float* __restrict__ cw3, const float* __restrict__ cb3,
                                     void* __restrict__ ws) {
  int i = blockIdx.x * blockDim.x + threadIdx.x;
  _Float16* H = (_Float16*)ws;
  float* BF = (float*)((char*)ws + H_TOTAL * 2);
  if (i < H_TOTAL) {
    float v = 0.f;
    if (i < 14336) {                       // wt1 [112][128], valid rows <97
      int n = i >> 7, k = i & 127;
      if (n < 97) v = w1[n * 128 + k];
    } else if (i < 28672) {                // wt2 [112][128], valid rows<97, cols<97
      int j = i - 14336; int n = j >> 7, k = j & 127;
      if (n < 97 && k < 97) v = w2[n * 97 + k];
    } else if (i < 34816) {                // wt3 [64][96]
      int j = i - 28672; v = w3[j];
    } else if (i < 38912) {                // cw1 [64][64]
      v = cw1[i - 34816];
    } else if (i < 43008) {                // cw2 [64][64]
      v = cw2[i - 38912];
    } else {                               // cw3 [16][64], valid rows<3
      int j = i - 43008; int n = j >> 6, k = j & 63;
      if (n < 3) v = cw3[n * 64 + k];
    }
    H[i] = (_Float16)v;
  } else if (i < H_TOTAL + F_TOTAL) {
    int j = i - H_TOTAL;
    float v = 0.f;
    if (j < 112)       { if (j < 97) v = b1[j]; }
    else if (j < 224)  { int t = j - 112; if (t < 97) v = b2[t]; }
    else if (j < 288)  v = b3[j - 224];
    else if (j < 352)  v = cb1[j - 288];
    else if (j < 416)  v = cb2[j - 352];
    else               { int t = j - 416; if (t < 3) v = cb3[t]; }
    BF[j] = v;
  }
}

// ---------------- fragment loaders (CDNA5 wave32 layouts) ----------------
// A 16x32 f16: lane holds row (lane&15); elems 0..15 = K kb+16*(lane>>4)+0..15
__device__ __forceinline__ v16h frag_a(const _Float16* scr, int lane, int kbase) {
  int r = lane & 15, hi = lane >> 4;
  const v8h* p0 = (const v8h*)(scr + r * 128 + kbase + hi * 16);
  const v8h* p1 = (const v8h*)(scr + r * 128 + kbase + 8 + hi * 16);
  v8h lo = *p0, hv = *p1;
  v16h a;
#pragma unroll
  for (int i = 0; i < 8; ++i) { a[i] = lo[i]; a[i + 8] = hv[i]; }
  return a;
}
// B 32x16 f16 from row-major W[n][k]: lane holds col (lane&15), 16 consecutive K from kbase+16*(lane>>4)
__device__ __forceinline__ v16h frag_b(const _Float16* wt, int stride, int lane, int ntile, int kbase) {
  int n = ntile * 16 + (lane & 15);
  int kb = kbase + ((lane >> 4) << 4);
  return *(const v16h*)(wt + (size_t)n * stride + kb);
}

template <int NT, int KT>
__device__ __forceinline__ void run_layer(v8f (&acc)[NT], const _Float16* scr,
                                          const _Float16* wt, int stride, int lane) {
#pragma unroll
  for (int nt = 0; nt < NT; ++nt) {
    v8f c = {0.f, 0.f, 0.f, 0.f, 0.f, 0.f, 0.f, 0.f};
#pragma unroll
    for (int kt = 0; kt < KT; ++kt) {
      v16h a = frag_a(scr, lane, kt * 32);
      v16h b = frag_b(wt, stride, lane, nt, kt * 32);
      c = __builtin_amdgcn_wmma_f32_16x16x32_f16(false, a, false, b, (short)0, c, false, false);
    }
    acc[nt] = c;
  }
}

// ---------------- fused triplane sample + MLP ----------------
__global__ __launch_bounds__(256) void triplane_main_kernel(
    const float* __restrict__ pts, const float* __restrict__ plane,
    const void* __restrict__ ws, float* __restrict__ out, int N) {
  __shared__ _Float16 s_scr[WAVES][16 * 128];
  __shared__ int s_idx[WAVES][16][12];
  __shared__ float s_w[WAVES][16][12];

  const int wave = threadIdx.x >> 5;
  const int lane = threadIdx.x & 31;
  const int ptbase = (blockIdx.x * WAVES + wave) * 16;
  _Float16* scr = s_scr[wave];

  const _Float16* W = (const _Float16*)ws;
  const _Float16* wt1 = W;
  const _Float16* wt2 = W + 14336;
  const _Float16* wt3 = W + 28672;
  const _Float16* cw1 = W + 34816;
  const _Float16* cw2 = W + 38912;
  const _Float16* cw3 = W + 43008;
  const float* BF = (const float*)((const char*)ws + H_TOTAL * 2);
  const float* b1 = BF;
  const float* b2 = BF + 112;
  const float* b3 = BF + 224;
  const float* cb1 = BF + 288;
  const float* cb2 = BF + 352;
  const float* cb3 = BF + 416;

  // ---- stage 1: per-point bilinear params (lanes 0..15) ----
  if (lane < 16) {
    int p = iminx(ptbase + lane, N - 1);
    float px = pts[p * 3 + 0], py = pts[p * 3 + 1], pz = pts[p * 3 + 2];
    float X = 2.f * px - 1.f, Y = 2.f * py - 1.f, Z = 2.f * pz - 1.f;
    float us[3] = {X, X, Y};
    float vs[3] = {Y, Z, Z};
#pragma unroll
    for (int s = 0; s < 3; ++s) {
      float fx = (us[s] + 1.f) * 127.5f, fy = (vs[s] + 1.f) * 127.5f;
      float x0f = floorf(fx), y0f = floorf(fy);
      float wx = fx - x0f, wy = fy - y0f;
      int ix0 = iminx(imaxx((int)x0f, 0), R_GRID - 1);
      int iy0 = iminx(imaxx((int)y0f, 0), R_GRID - 1);
      int ix1 = iminx(ix0 + 1, R_GRID - 1);
      int iy1 = iminx(iy0 + 1, R_GRID - 1);
      s_idx[wave][lane][s * 4 + 0] = iy0 * R_GRID + ix0;
      s_idx[wave][lane][s * 4 + 1] = iy0 * R_GRID + ix1;
      s_idx[wave][lane][s * 4 + 2] = iy1 * R_GRID + ix0;
      s_idx[wave][lane][s * 4 + 3] = iy1 * R_GRID + ix1;
      s_w[wave][lane][s * 4 + 0] = (1.f - wx) * (1.f - wy);
      s_w[wave][lane][s * 4 + 1] = wx * (1.f - wy);
      s_w[wave][lane][s * 4 + 2] = (1.f - wx) * wy;
      s_w[wave][lane][s * 4 + 3] = wx * wy;
    }
  }
  __builtin_amdgcn_wave_barrier();  // same-wave DS ops are in-order; barrier blocks sched reorder

  // ---- stage 2: gather 16x128 feature tile -> LDS (f16, A staging) ----
#pragma unroll 1
  for (int p = 0; p < 16; ++p) {
    int id[12]; float w[12];
#pragma unroll
    for (int j = 0; j < 12; ++j) { id[j] = s_idx[wave][p][j]; w[j] = s_w[wave][p][j]; }
#pragma unroll
    for (int t = 0; t < 4; ++t) {
      int c = lane + t * 32;
      const float* pc = plane + (size_t)c * (R_GRID * R_GRID);
      float f = 1.f;
#pragma unroll
      for (int s = 0; s < 3; ++s) {
        float v = w[s * 4 + 0] * pc[id[s * 4 + 0]] + w[s * 4 + 1] * pc[id[s * 4 + 1]] +
                  w[s * 4 + 2] * pc[id[s * 4 + 2]] + w[s * 4 + 3] * pc[id[s * 4 + 3]];
        f *= v;
      }
      scr[p * 128 + c] = (_Float16)f;
    }
  }

  const int col = lane & 15;
  const int m0 = (lane >> 4) * 8;
  size_t Ns = (size_t)N;
  float* out_rgb = out;
  float* out_den = out + 3 * Ns;
  float* out_sem = out + 4 * Ns;
  float* out_geo = out + 68 * Ns;

  // ---- layer 1: h1 = relu(feat @ w1.T + b1) -> scratch [16][112], pad K to 128 ----
  {
    v8f acc[7];
    run_layer<7, 4>(acc, scr, wt1, 128, lane);
#pragma unroll
    for (int nt = 0; nt < 7; ++nt) {
      int n = nt * 16 + col;
      float bv = b1[n];
#pragma unroll
      for (int v = 0; v < 8; ++v) {
        float h = acc[nt][v] + bv;
        h = h > 0.f ? h : 0.f;
        scr[(m0 + v) * 128 + n] = (_Float16)h;
      }
    }
#pragma unroll
    for (int v = 0; v < 8; ++v) scr[(m0 + v) * 128 + 112 + col] = (_Float16)0.f;  // zero K-pad
  }

  // ---- layer 2: h = h1 @ w2.T + b2 ; geo = h[:,:96], density = exp(h[:,96]) ----
  {
    v8f acc[7];
    run_layer<7, 4>(acc, scr, wt2, 128, lane);
#pragma unroll
    for (int nt = 0; nt < 7; ++nt) {
      int n = nt * 16 + col;
      float bv = b2[n];
#pragma unroll
      for (int v = 0; v < 8; ++v) {
        int pt = ptbase + m0 + v;
        float h = acc[nt][v] + bv;
        if (n < 96) {
          if (pt < N) out_geo[(size_t)pt * 96 + n] = h;
          float r = h > 0.f ? h : 0.f;
          scr[(m0 + v) * 128 + n] = (_Float16)r;  // relu(geo) -> A for layer 3 (K=96)
        } else if (n == 96) {
          if (pt < N) out_den[pt] = __expf(h);
        }
      }
    }
  }

  // ---- layer 3: sem = relu(geo) @ w3.T + b3 -> [16][64] ----
  {
    v8f acc[4];
    run_layer<4, 3>(acc, scr, wt3, 96, lane);
#pragma unroll
    for (int nt = 0; nt < 4; ++nt) {
      int n = nt * 16 + col;
      float bv = b3[n];
#pragma unroll
      for (int v = 0; v < 8; ++v) {
        int pt = ptbase + m0 + v;
        float sm = acc[nt][v] + bv;
        if (pt < N) out_sem[(size_t)pt * 64 + n] = sm;
        scr[(m0 + v) * 128 + n] = (_Float16)sm;  // raw sem -> A for layer 4
      }
    }
  }

  // ---- layer 4: c = relu(sem @ cw1.T + cb1) ----
  {
    v8f acc[4];
    run_layer<4, 2>(acc, scr, cw1, 64, lane);
#pragma unroll
    for (int nt = 0; nt < 4; ++nt) {
      int n = nt * 16 + col;
      float bv = cb1[n];
#pragma unroll
      for (int v = 0; v < 8; ++v) {
        float h = acc[nt][v] + bv;
        h = h > 0.f ? h : 0.f;
        scr[(m0 + v) * 128 + n] = (_Float16)h;
      }
    }
  }

  // ---- layer 5: c = relu(c @ cw2.T + cb2) ----
  {
    v8f acc[4];
    run_layer<4, 2>(acc, scr, cw2, 64, lane);
#pragma unroll
    for (int nt = 0; nt < 4; ++nt) {
      int n = nt * 16 + col;
      float bv = cb2[n];
#pragma unroll
      for (int v = 0; v < 8; ++v) {
        float h = acc[nt][v] + bv;
        h = h > 0.f ? h : 0.f;
        scr[(m0 + v) * 128 + n] = (_Float16)h;
      }
    }
  }

  // ---- layer 6: rgb = sigmoid(c @ cw3.T + cb3), 3 valid cols ----
  {
    v8f acc[1];
    run_layer<1, 2>(acc, scr, cw3, 64, lane);
    if (col < 3) {
      float bv = cb3[col];
#pragma unroll
      for (int v = 0; v < 8; ++v) {
        int pt = ptbase + m0 + v;
        float x = acc[0][v] + bv;
        float sg = 1.f / (1.f + __expf(-x));
        if (pt < N) out_rgb[(size_t)pt * 3 + col] = sg;
      }
    }
  }
}

extern "C" void kernel_launch(void* const* d_in, const int* in_sizes, int n_in,
                              void* d_out, int out_size, void* d_ws, size_t ws_size,
                              hipStream_t stream) {
  const float* pts = (const float*)d_in[0];
  const float* plane = (const float*)d_in[1];
  const float* w1 = (const float*)d_in[2];
  const float* b1 = (const float*)d_in[3];
  const float* w2 = (const float*)d_in[4];
  const float* b2 = (const float*)d_in[5];
  const float* w3 = (const float*)d_in[6];
  const float* b3 = (const float*)d_in[7];
  const float* cw1 = (const float*)d_in[8];
  const float* cb1 = (const float*)d_in[9];
  const float* cw2 = (const float*)d_in[10];
  const float* cb2 = (const float*)d_in[11];
  const float* cw3 = (const float*)d_in[12];
  const float* cb3 = (const float*)d_in[13];

  int N = in_sizes[0] / 3;  // 2048*512 points

  int prep_elems = H_TOTAL + F_TOTAL;
  triplane_prep_kernel<<<(prep_elems + 255) / 256, 256, 0, stream>>>(
      w1, b1, w2, b2, w3, b3, cw1, cb1, cw2, cb2, cw3, cb3, d_ws);

  int blocks = (N + 16 * WAVES - 1) / (16 * WAVES);
  triplane_main_kernel<<<blocks, 256, 0, stream>>>(pts, plane, d_ws, (float*)d_out, N);
}
